// Money_former_MLA_DINT_cog_attn_MTP_3813930959352
// MI455X (gfx1250) — matrix operand order repaired and checked by reading, not compiled, not run on hardware
//
#include <hip/hip_runtime.h>
#include <hip/hip_bf16.h>

typedef __attribute__((ext_vector_type(16))) __bf16 v16bf;
typedef __attribute__((ext_vector_type(8)))  __bf16 v8bf;
typedef __attribute__((ext_vector_type(8)))  float  v8f;
typedef __attribute__((ext_vector_type(4)))  float  v4f;

constexpr int   kB  = 2;
constexpr int   kS  = 1025;   // NSEQ*SPS + 1
constexpr int   kSP = 1056;   // S padded to multiple of 32 (zero-filled tail)
constexpr float kScale      = 1.0f / 48.0f;   // 1/((64+32)/2)
constexpr float kLambdaInit = 0.2f;           // 0.8 - 0.6*exp(0)
constexpr float kEpsRms     = 1.1920929e-07f;
constexpr float kEpsHead    = 1e-05f;

// ---- WMMA fragment layout (CDNA5 ISA 7.12.2) ----
// A 16x32 bf16 : lane<16 row M=lane, elems 0..7 -> K k0+8h+0..7, elems 8..15 -> K k0+16+8h+0..7
// B 32x16 bf16 : lanes n=lane%16, same order as A; this code loads A and B both as K k0+16h+0..15
// C/D          : row r + 8*(lane/16), col lane%16

__device__ __forceinline__ v16bf ldfrag(const __bf16* p0, const __bf16* p1) {
  v8bf lo = *(const v8bf*)p0;
  v8bf hi = *(const v8bf*)p1;
  v16bf r;
#pragma unroll
  for (int e = 0; e < 8; e++) { r[e] = lo[e]; r[8 + e] = hi[e]; }
  return r;
}

__device__ __forceinline__ v8f wmma_bf16(const v16bf& a, const v16bf& b, const v8f& c) {
  return __builtin_amdgcn_wmma_f32_16x16x32_bf16(false, a, false, b, (short)0, c, false, false);
}

// ---------------------------------------------------------------------------
// fp32 -> bf16 bulk convert (n % 8 == 0)
// ---------------------------------------------------------------------------
__global__ void cvt_bf16_kernel(const float* __restrict__ in, __bf16* __restrict__ out, int n) {
  const int i = (blockIdx.x * blockDim.x + threadIdx.x) * 8;
  if (i >= n) return;
  v4f a = *(const v4f*)(in + i);
  v4f b = *(const v4f*)(in + i + 4);
  v8bf o;
#pragma unroll
  for (int e = 0; e < 4; e++) { o[e] = (__bf16)a[e]; o[4 + e] = (__bf16)b[e]; }
  *(v8bf*)(out + i) = o;
}

// ---------------------------------------------------------------------------
// Weight transpose + convert: WT[n][k] = (bf16) W[k][n]
// ---------------------------------------------------------------------------
__global__ void wtrans_kernel(const float* __restrict__ W, __bf16* __restrict__ WT, int K, int N) {
  const int idx = blockIdx.x * blockDim.x + threadIdx.x;
  if (idx >= N * K) return;
  const int n = idx / K, k = idx % K;
  WT[idx] = (__bf16)W[(size_t)k * N + n];
}

// ---------------------------------------------------------------------------
// GEMM: C[M,N] f32 = A[M,K] bf16 @ BT[N,K] bf16 (B pre-transposed).
// One wave -> 16x32 tile; k-step = 6 x b128 loads + 2 WMMA. K%32==0, N%32==0.
// ---------------------------------------------------------------------------
__global__ __launch_bounds__(32) void gemm_bf16_kernel(
    const __bf16* __restrict__ A, const __bf16* __restrict__ BT,
    float* __restrict__ C, int M, int N, int K)
{
  const int tn = blockIdx.x * 32;
  const int tm = blockIdx.y * 16;
  const int lane = threadIdx.x;
  const int half = lane >> 4;
  const int l16  = lane & 15;

  const int arow = min(tm + l16, M - 1);           // clamp: rows >= M corrupt only unsaved rows
  const __bf16* ap  = A  + (size_t)arow * K;
  const __bf16* b0p = BT + (size_t)(tn + l16) * K;
  const __bf16* b1p = BT + (size_t)(tn + 16 + l16) * K;

  v8f acc0 = {}, acc1 = {};
  for (int k0 = 0; k0 < K; k0 += 32) {
    __builtin_prefetch(ap + k0 + 128, 0, 1);       // global_prefetch_b8
    v16bf a  = ldfrag(ap  + k0 + 16 * half, ap  + k0 + 16 * half + 8);
    v16bf b0 = ldfrag(b0p + k0 + 16 * half, b0p + k0 + 16 * half + 8);
    v16bf b1 = ldfrag(b1p + k0 + 16 * half, b1p + k0 + 16 * half + 8);
    acc0 = wmma_bf16(a, b0, acc0);
    acc1 = wmma_bf16(a, b1, acc1);
  }
#pragma unroll
  for (int r = 0; r < 8; r++) {
    const int row = tm + r + (half << 3);
    if (row < M) {
      C[(size_t)row * N + tn + l16]      = acc0[r];
      C[(size_t)row * N + tn + 16 + l16] = acc1[r];
    }
  }
}

// ---------------------------------------------------------------------------
// RMS norm, bf16 output: out[row][j] = in[row][j] * rsqrt(mean(x^2)+eps) * g[j]
// ---------------------------------------------------------------------------
__global__ __launch_bounds__(32) void rms_kernel(
    const float* __restrict__ in, const float* __restrict__ g,
    __bf16* __restrict__ out, int width, int in_stride, float eps)
{
  const int row  = blockIdx.x;
  const int lane = threadIdx.x;
  const float* ip = in + (size_t)row * in_stride;
  float ss = 0.0f;
  for (int j = lane; j < width; j += 32) { float v = ip[j]; ss += v * v; }
#pragma unroll
  for (int off = 16; off >= 1; off >>= 1) ss += __shfl_xor(ss, off, 32);
  const float inv = rsqrtf(ss / (float)width + eps);
  __bf16* op = out + (size_t)row * width;
  for (int j = lane; j < width; j += 32) op[j] = (__bf16)(ip[j] * inv * g[j]);
}

// ---------------------------------------------------------------------------
// Assemble padded bf16 Q/K (RoPE applied, 48 dims zero-padded to 64) and V^T.
//  qbf/kbf: [B][32][kSP][64], vT: [B][16][64][kSP]; pad region zero-filled.
// one thread per (b, s_padded, subhead)
// ---------------------------------------------------------------------------
__global__ void assemble_kernel(
    const float* __restrict__ ckv,   // [B*S][288]  (k_rope = cols 256..287)
    const float* __restrict__ kv,    // [B*S][2048] (per head: k 0..63, v 64..127)
    const float* __restrict__ qf,    // [B*S][1536] (per head: q 0..63, q_rope 64..95)
    const float* __restrict__ fcos,  // [256][8]
    const float* __restrict__ fsin,  // [256][8]
    __bf16* __restrict__ qbf, __bf16* __restrict__ kbf, __bf16* __restrict__ vT)
{
  const int idx = blockIdx.x * blockDim.x + threadIdx.x;
  if (idx >= kB * kSP * 32) return;
  const int h2  = idx & 31;
  const int bsp = idx >> 5;
  const int b   = bsp / kSP;
  const int s   = bsp % kSP;
  const int h   = h2 >> 1;
  const int sub = h2 & 1;

  const size_t dq = ((size_t)(b * 32 + h2) * kSP + s) * 64;
  const size_t vtb = (size_t)(b * 16 + h) * 64 * kSP + s;

  if (s >= kS) {  // zero the padding tail
    v8bf z = {};
#pragma unroll
    for (int t = 0; t < 8; t++) { *(v8bf*)(qbf + dq + 8 * t) = z; *(v8bf*)(kbf + dq + 8 * t) = z; }
    if (sub == 0) {
#pragma unroll
      for (int j = 0; j < 64; j++) vT[vtb + (size_t)j * kSP] = (__bf16)0.0f;
    }
    return;
  }

  const int bs = b * kS + s;
  float cs[8], sn[8];
  if (s == 0) {
#pragma unroll
    for (int i = 0; i < 8; i++) { cs[i] = 1.0f; sn[i] = 0.0f; }
  } else {
    const int pos = (s - 1) & 255;
    v4f c0 = *(const v4f*)(fcos + pos * 8), c1 = *(const v4f*)(fcos + pos * 8 + 4);
    v4f s0 = *(const v4f*)(fsin + pos * 8), s1 = *(const v4f*)(fsin + pos * 8 + 4);
#pragma unroll
    for (int i = 0; i < 4; i++) { cs[i] = c0[i]; cs[4 + i] = c1[i]; sn[i] = s0[i]; sn[4 + i] = s1[i]; }
  }

  __bf16 row[64];

  // ---- Q ----
  {
    const v4f* qn = (const v4f*)(qf + (size_t)bs * 1536 + h * 96 + sub * 32);
#pragma unroll
    for (int t = 0; t < 8; t++) { v4f v = qn[t];
#pragma unroll
      for (int e = 0; e < 4; e++) row[4 * t + e] = (__bf16)v[e]; }
    const v4f* qr = (const v4f*)(qf + (size_t)bs * 1536 + h * 96 + 64 + sub * 16);
    float xr[16];
#pragma unroll
    for (int t = 0; t < 4; t++) { v4f v = qr[t];
#pragma unroll
      for (int e = 0; e < 4; e++) xr[4 * t + e] = v[e]; }
#pragma unroll
    for (int i = 0; i < 8; i++) {
      row[32 + 2 * i]     = (__bf16)(xr[2 * i] * cs[i] - xr[2 * i + 1] * sn[i]);
      row[32 + 2 * i + 1] = (__bf16)(xr[2 * i] * sn[i] + xr[2 * i + 1] * cs[i]);
    }
#pragma unroll
    for (int j = 48; j < 64; j++) row[j] = (__bf16)0.0f;
#pragma unroll
    for (int t = 0; t < 8; t++) *(v8bf*)(qbf + dq + 8 * t) = *(v8bf*)(row + 8 * t);
  }

  // ---- K ----
  {
    const v4f* kn = (const v4f*)(kv + (size_t)bs * 2048 + h * 128 + sub * 32);
#pragma unroll
    for (int t = 0; t < 8; t++) { v4f v = kn[t];
#pragma unroll
      for (int e = 0; e < 4; e++) row[4 * t + e] = (__bf16)v[e]; }
    const v4f* kr = (const v4f*)(ckv + (size_t)bs * 288 + 256 + sub * 16);
    float xr[16];
#pragma unroll
    for (int t = 0; t < 4; t++) { v4f v = kr[t];
#pragma unroll
      for (int e = 0; e < 4; e++) xr[4 * t + e] = v[e]; }
#pragma unroll
    for (int i = 0; i < 8; i++) {
      row[32 + 2 * i]     = (__bf16)(xr[2 * i] * cs[i] - xr[2 * i + 1] * sn[i]);
      row[32 + 2 * i + 1] = (__bf16)(xr[2 * i] * sn[i] + xr[2 * i + 1] * cs[i]);
    }
#pragma unroll
    for (int j = 48; j < 64; j++) row[j] = (__bf16)0.0f;
#pragma unroll
    for (int t = 0; t < 8; t++) *(v8bf*)(kbf + dq + 8 * t) = *(v8bf*)(row + 8 * t);
  }

  // ---- V^T (once per full head) ----
  if (sub == 0) {
    const v4f* vs = (const v4f*)(kv + (size_t)bs * 2048 + h * 128 + 64);
#pragma unroll
    for (int t = 0; t < 16; t++) { v4f v = vs[t];
#pragma unroll
      for (int e = 0; e < 4; e++) vT[vtb + (size_t)(4 * t + e) * kSP] = (__bf16)v[e]; }
  }
}

// mask==0 (visible) iff k==0, or (q>=1 && ((k-1)%256) <= ((q-1)%256))
__device__ __forceinline__ bool visible(int q, int k) {
  if (k == 0) return true;
  if (q == 0) return false;
  return ((k - 1) & 255) <= ((q - 1) & 255);
}

// ---------------------------------------------------------------------------
// Differential attention, flash-style, one wave per (b, head, 16-query tile).
// Pass 1: per-lane online stats (max, Z, signed-Z) over the lane's columns,
//         single shfl_xor butterfly merge at the end.
// Pass 2: recompute scores, combined weight a = p0 - lam*p1 + G*lam staged in
//         bf16 LDS tile, fed as A-fragment into PV WMMA against V^T.
// Then per-head RMS * g_head, bf16 store to att_out[B*S][1024].
// ---------------------------------------------------------------------------
__global__ __launch_bounds__(32) void attn_kernel(
    const __bf16* __restrict__ qbf, const __bf16* __restrict__ kbf,
    const __bf16* __restrict__ vT,
    const float* __restrict__ g_head,
    const float* __restrict__ lq1, const float* __restrict__ lk1,
    const float* __restrict__ lq2, const float* __restrict__ lk2,
    __bf16* __restrict__ att_out)
{
  const int qt   = blockIdx.x;   // 0..64
  const int h    = blockIdx.y;   // 0..15
  const int b    = blockIdx.z;   // 0..1
  const int lane = threadIdx.x;
  const int half = lane >> 4;
  const int l16  = lane & 15;

  __shared__ __bf16 stg[16][32];

  // lambda scalar (HD/2 == 32 == wave32)
  float t1 = lq1[lane] * lk1[lane];
  float t2 = lq2[lane] * lk2[lane];
#pragma unroll
  for (int off = 16; off >= 1; off >>= 1) {
    t1 += __shfl_xor(t1, off, 32);
    t2 += __shfl_xor(t2, off, 32);
  }
  const float lam = __expf(t1) - __expf(t2) + kLambdaInit;

  const __bf16* q0base = qbf + (size_t)(b * 32 + 2 * h)     * kSP * 64;
  const __bf16* q1base = qbf + (size_t)(b * 32 + 2 * h + 1) * kSP * 64;
  const __bf16* k0base = kbf + (size_t)(b * 32 + 2 * h)     * kSP * 64;
  const __bf16* k1base = kbf + (size_t)(b * 32 + 2 * h + 1) * kSP * 64;
  const __bf16* vTbase = vT  + (size_t)(b * 16 + h) * 64 * kSP;

  // Q fragments (K-dim 0..63 -> two 32-chunks per sub-head); rows >= kS are zero rows
  const int qrow = qt * 16 + l16;   // < 1040 < kSP, always in bounds
  const __bf16* qp0 = q0base + (size_t)qrow * 64;
  const __bf16* qp1 = q1base + (size_t)qrow * 64;
  const v16bf q0a = ldfrag(qp0 + 16 * half,      qp0 + 16 * half + 8);
  const v16bf q0b = ldfrag(qp0 + 32 + 16 * half, qp0 + 40 + 16 * half);
  const v16bf q1a = ldfrag(qp1 + 16 * half,      qp1 + 16 * half + 8);
  const v16bf q1b = ldfrag(qp1 + 32 + 16 * half, qp1 + 40 + 16 * half);

  // ---- Pass 1: per-lane online softmax stats ----
  float m0l[8], Z0l[8], SGl[8], m1l[8], Z1l[8];
#pragma unroll
  for (int r = 0; r < 8; r++) { m0l[r] = -1e30f; Z0l[r] = 0.f; SGl[r] = 0.f; m1l[r] = -1e30f; Z1l[r] = 0.f; }

#pragma unroll 1
  for (int kt = 0; kt < 65; kt++) {
    const int kcol = kt * 16 + l16;
    const __bf16* kp0 = k0base + (size_t)kcol * 64;
    const __bf16* kp1 = k1base + (size_t)kcol * 64;
    v16bf b0a = ldfrag(kp0 + 16 * half,      kp0 + 16 * half + 8);
    v16bf b0b = ldfrag(kp0 + 32 + 16 * half, kp0 + 40 + 16 * half);
    v16bf b1a = ldfrag(kp1 + 16 * half,      kp1 + 16 * half + 8);
    v16bf b1b = ldfrag(kp1 + 32 + 16 * half, kp1 + 40 + 16 * half);
    v8f s0 = {}, s1 = {};
    s0 = wmma_bf16(q0a, b0a, s0); s0 = wmma_bf16(q0b, b0b, s0);
    s1 = wmma_bf16(q1a, b1a, s1); s1 = wmma_bf16(q1b, b1b, s1);

#pragma unroll
    for (int r = 0; r < 8; r++) {
      const int row = qt * 16 + r + (half << 3);
      if ((kcol < kS) && visible(row, kcol)) {
        const float raw0 = s0[r] * kScale;
        const float raw1 = s1[r] * kScale;
        const float av0 = fabsf(raw0), av1 = fabsf(raw1);
        const float mn0 = fmaxf(m0l[r], av0);
        const float mn1 = fmaxf(m1l[r], av1);
        const float c0 = __expf(m0l[r] - mn0), e0 = __expf(av0 - mn0);
        const float c1 = __expf(m1l[r] - mn1), e1 = __expf(av1 - mn1);
        Z0l[r] = Z0l[r] * c0 + e0;
        SGl[r] = SGl[r] * c0 + (raw0 > 0.f ? e0 : (raw0 < 0.f ? -e0 : 0.f));
        m0l[r] = mn0;
        Z1l[r] = Z1l[r] * c1 + e1;
        m1l[r] = mn1;
      }
    }
  }

  // butterfly merge across the 16-lane group
  float m0[8], m1[8], iZ0[8], iZ1[8], G[8];
#pragma unroll
  for (int r = 0; r < 8; r++) {
    float ma = m0l[r], Za = Z0l[r], Sa = SGl[r];
    float mb = m1l[r], Zb = Z1l[r];
#pragma unroll
    for (int off = 8; off >= 1; off >>= 1) {
      float mo = __shfl_xor(ma, off, 32), Zo = __shfl_xor(Za, off, 32), So = __shfl_xor(Sa, off, 32);
      float mn = fmaxf(ma, mo);
      float ca = __expf(ma - mn), co = __expf(mo - mn);
      Za = Za * ca + Zo * co; Sa = Sa * ca + So * co; ma = mn;
      float no = __shfl_xor(mb, off, 32), Yo = __shfl_xor(Zb, off, 32);
      float nn = fmaxf(mb, no);
      Zb = Zb * __expf(mb - nn) + Yo * __expf(no - nn); mb = nn;
    }
    m0[r] = ma; iZ0[r] = 1.0f / Za;
    m1[r] = mb; iZ1[r] = 1.0f / Zb;
    G[r] = Sa / Za * (1.0f / (float)kS);   // row-mean of signed a0
  }

  // ---- Pass 2: recompute scores, combine, PV via WMMA ----
  v8f o[4] = {{}, {}, {}, {}};
#pragma unroll 1
  for (int kc = 0; kc < 33; kc++) {   // 33 chunks of 32 keys = kSP
#pragma unroll
    for (int t2 = 0; t2 < 2; t2++) {
      const int kcol = kc * 32 + t2 * 16 + l16;   // < kSP, in bounds (pad rows are zero)
      const __bf16* kp0 = k0base + (size_t)kcol * 64;
      const __bf16* kp1 = k1base + (size_t)kcol * 64;
      v16bf b0a = ldfrag(kp0 + 16 * half,      kp0 + 16 * half + 8);
      v16bf b0b = ldfrag(kp0 + 32 + 16 * half, kp0 + 40 + 16 * half);
      v16bf b1a = ldfrag(kp1 + 16 * half,      kp1 + 16 * half + 8);
      v16bf b1b = ldfrag(kp1 + 32 + 16 * half, kp1 + 40 + 16 * half);
      v8f s0 = {}, s1 = {};
      s0 = wmma_bf16(q0a, b0a, s0); s0 = wmma_bf16(q0b, b0b, s0);
      s1 = wmma_bf16(q1a, b1a, s1); s1 = wmma_bf16(q1b, b1b, s1);
#pragma unroll
      for (int r = 0; r < 8; r++) {
        const int row = qt * 16 + r + (half << 3);
        float a = 0.0f;
        if ((kcol < kS) && visible(row, kcol)) {
          const float raw0 = s0[r] * kScale;
          const float raw1 = s1[r] * kScale;
          const float e0 = __expf(fabsf(raw0) - m0[r]);
          const float e1 = __expf(fabsf(raw1) - m1[r]);
          const float p0 = (raw0 > 0.f ? e0 : (raw0 < 0.f ? -e0 : 0.f)) * iZ0[r];
          const float p1 = (raw1 > 0.f ? e1 : (raw1 < 0.f ? -e1 : 0.f)) * iZ1[r];
          a = p0 - lam * p1 + G[r] * lam;
        }
        stg[r + (half << 3)][t2 * 16 + l16] = (__bf16)a;
      }
    }
    __syncthreads();

    const v16bf af = ldfrag(&stg[l16][16 * half], &stg[l16][16 * half + 8]);
#pragma unroll
    for (int j = 0; j < 4; j++) {
      const __bf16* vp = vTbase + (size_t)(j * 16 + l16) * kSP + kc * 32 + 16 * half;
      v16bf bv = ldfrag(vp, vp + 8);
      o[j] = wmma_bf16(af, bv, o[j]);
    }
    __syncthreads();
  }

  // ---- per-head RMS * gamma, bf16 store ----
  float ghv[4];
#pragma unroll
  for (int j = 0; j < 4; j++) ghv[j] = g_head[j * 16 + l16];

#pragma unroll
  for (int r = 0; r < 8; r++) {
    float ss = o[0][r] * o[0][r] + o[1][r] * o[1][r] + o[2][r] * o[2][r] + o[3][r] * o[3][r];
#pragma unroll
    for (int off = 8; off >= 1; off >>= 1) ss += __shfl_xor(ss, off, 32);
    const float inv = rsqrtf(ss * (1.0f / 64.0f) + kEpsHead);
    const int row = qt * 16 + r + (half << 3);
    if (row < kS) {
      const size_t base = ((size_t)b * kS + row) * 1024 + h * 64;
#pragma unroll
      for (int j = 0; j < 4; j++)
        att_out[base + j * 16 + l16] = (__bf16)(o[j][r] * inv * ghv[j]);
    }
  }
}

// ---------------------------------------------------------------------------
extern "C" void kernel_launch(void* const* d_in, const int* in_sizes, int n_in,
                              void* d_out, int out_size, void* d_ws, size_t ws_size,
                              hipStream_t stream) {
  const float* x    = (const float*)d_in[0];
  // d_in[1] = mask (int32) — computed procedurally
  const float* fcos = (const float*)d_in[2];
  const float* fsin = (const float*)d_in[3];
  const float* Wkvd = (const float*)d_in[4];
  const float* Wqd  = (const float*)d_in[5];
  const float* Wkvu = (const float*)d_in[6];
  const float* Wqu  = (const float*)d_in[7];
  const float* Wo   = (const float*)d_in[8];
  const float* gkv  = (const float*)d_in[9];
  const float* gq   = (const float*)d_in[10];
  const float* gh   = (const float*)d_in[11];
  const float* lq1  = (const float*)d_in[12];
  const float* lk1  = (const float*)d_in[13];
  const float* lq2  = (const float*)d_in[14];
  const float* lk2  = (const float*)d_in[15];

  const int BS = kB * kS;  // 2050

  char* wsp = (char*)d_ws;
  auto alloc = [&](size_t bytes) -> void* {
    void* p = (void*)wsp;
    wsp += (bytes + 255) & ~(size_t)255;
    return p;
  };
  __bf16* xbf    = (__bf16*)alloc((size_t)BS * 1024 * 2);
  __bf16* WkvdT  = (__bf16*)alloc((size_t)288  * 1024 * 2);
  __bf16* WqdT   = (__bf16*)alloc((size_t)384  * 1024 * 2);
  __bf16* WkvuT  = (__bf16*)alloc((size_t)2048 * 256  * 2);
  __bf16* WquT   = (__bf16*)alloc((size_t)1536 * 384  * 2);
  __bf16* WoT    = (__bf16*)alloc((size_t)1024 * 1024 * 2);
  float*  ckv    = (float*) alloc((size_t)BS * 288 * 4);
  float*  qd     = (float*) alloc((size_t)BS * 384 * 4);
  __bf16* ckvn   = (__bf16*)alloc((size_t)BS * 256 * 2);
  __bf16* qdn    = (__bf16*)alloc((size_t)BS * 384 * 2);
  float*  kvu    = (float*) alloc((size_t)BS * 2048 * 4);
  float*  qfu    = (float*) alloc((size_t)BS * 1536 * 4);
  __bf16* qbf    = (__bf16*)alloc((size_t)kB * 32 * kSP * 64 * 2);
  __bf16* kbf    = (__bf16*)alloc((size_t)kB * 32 * kSP * 64 * 2);
  __bf16* vT     = (__bf16*)alloc((size_t)kB * 16 * 64 * kSP * 2);
  __bf16* attout = (__bf16*)alloc((size_t)BS * 1024 * 2);

  const int mt = (BS + 15) / 16;  // 129 row tiles
  dim3 wave(32);

  // 0) convert / transpose
  cvt_bf16_kernel<<<(BS * 1024 / 8 + 255) / 256, 256, 0, stream>>>(x, xbf, BS * 1024);
  wtrans_kernel<<<(288  * 1024 + 255) / 256, 256, 0, stream>>>(Wkvd, WkvdT, 1024, 288);
  wtrans_kernel<<<(384  * 1024 + 255) / 256, 256, 0, stream>>>(Wqd,  WqdT,  1024, 384);
  wtrans_kernel<<<(2048 * 256  + 255) / 256, 256, 0, stream>>>(Wkvu, WkvuT, 256,  2048);
  wtrans_kernel<<<(1536 * 384  + 255) / 256, 256, 0, stream>>>(Wqu,  WquT,  384,  1536);
  wtrans_kernel<<<(1024 * 1024 + 255) / 256, 256, 0, stream>>>(Wo,   WoT,   1024, 1024);

  // 1) down projections
  gemm_bf16_kernel<<<dim3(288 / 32, mt), wave, 0, stream>>>(xbf, WkvdT, ckv, BS, 288, 1024);
  gemm_bf16_kernel<<<dim3(384 / 32, mt), wave, 0, stream>>>(xbf, WqdT,  qd,  BS, 384, 1024);
  // 2) RMS norms (bf16 out)
  rms_kernel<<<BS, wave, 0, stream>>>(ckv, gkv, ckvn, 256, 288, kEpsRms);
  rms_kernel<<<BS, wave, 0, stream>>>(qd,  gq,  qdn,  384, 384, kEpsRms);
  // 3) up projections
  gemm_bf16_kernel<<<dim3(2048 / 32, mt), wave, 0, stream>>>(ckvn, WkvuT, kvu, BS, 2048, 256);
  gemm_bf16_kernel<<<dim3(1536 / 32, mt), wave, 0, stream>>>(qdn,  WquT,  qfu, BS, 1536, 384);
  // 4) RoPE + assemble padded bf16 Q/K/V^T
  {
    const int n = kB * kSP * 32;
    assemble_kernel<<<(n + 127) / 128, 128, 0, stream>>>(ckv, kvu, qfu, fcos, fsin, qbf, kbf, vT);
  }
  // 5) differential attention + per-head RMS
  attn_kernel<<<dim3(65, 16, kB), wave, 0, stream>>>(qbf, kbf, vT, gh, lq1, lk1, lq2, lk2, attout);
  // 6) output projection
  gemm_bf16_kernel<<<dim3(1024 / 32, mt), wave, 0, stream>>>(attout, WoT, (float*)d_out, BS, 1024, 1024);

  (void)in_sizes; (void)n_in; (void)out_size; (void)ws_size;
}
